// TanhAttn_592705487555
// MI455X (gfx1250) — matrix operand order repaired and checked by reading, not compiled, not run on hardware
//
#include <hip/hip_runtime.h>
#include <hip/hip_bf16.h>
#include <math.h>

#define S_LEN 2048
#define B_SZ  16
#define D_DIM 512
#define KB    32   // keys per block in attention
#define QB    32   // query rows per workgroup

typedef __attribute__((ext_vector_type(16))) __bf16 v16bf;
typedef __attribute__((ext_vector_type(8)))  float  v8f;
typedef __attribute__((ext_vector_type(4)))  unsigned int v4u;
typedef __attribute__((ext_vector_type(8)))  unsigned int v8u;

union Frag16 { v16bf v; uint4 q[2]; };           // 16 bf16 = one WMMA A/B fragment
union Acc8   { v8f v; float f[8]; };             // 16x16 f32 C/D fragment
union U4B    { uint4 q; __bf16 b[8]; };

// ---------------------------------------------------------------------------
// seq [S,B,D] f32  ->  abf [B,S,D] bf16   (transpose + downconvert)
// ---------------------------------------------------------------------------
__global__ __launch_bounds__(256) void cvt_seq(const float* __restrict__ seq,
                                               __bf16* __restrict__ abf) {
  size_t idx = (size_t)blockIdx.x * blockDim.x + threadIdx.x;
  size_t total = (size_t)S_LEN * B_SZ * D_DIM;
  if (idx >= total) return;
  int d = (int)(idx % D_DIM);
  size_t r = idx / D_DIM;            // r = b*S + s
  int s = (int)(r % S_LEN);
  int b = (int)(r / S_LEN);
  abf[idx] = (__bf16)seq[((size_t)s * B_SZ + b) * D_DIM + d];
}

__global__ __launch_bounds__(256) void cvt_w(const float* __restrict__ w,
                                             __bf16* __restrict__ wbf, int n) {
  int i = blockIdx.x * blockDim.x + threadIdx.x;
  if (i < n) wbf[i] = (__bf16)w[i];
}

// ---------------------------------------------------------------------------
// value = tanh(A @ Wv^T + bv),  A = abf [M=B*S, K=D], Wv [N=D, K=D] row-major
// one wave -> 16x64 output tile; fragment loads double-buffered so WMMAs on
// buffer `cur` overlap the global loads for buffer `cur^1`.
// ---------------------------------------------------------------------------
__global__ __launch_bounds__(128) void value_gemm(const __bf16* __restrict__ abf,
                                                  const __bf16* __restrict__ wbf,
                                                  const float*  __restrict__ bv,
                                                  __bf16* __restrict__ vbf) {
  const int lane = threadIdx.x & 31;
  const int wid  = threadIdx.x >> 5;
  const int gw   = blockIdx.x * 4 + wid;     // global wave id
  const int mt   = gw >> 3;                  // m-tile (16 rows), 0..2047
  const int ng   = gw & 7;                   // 64-col group
  const int m    = lane & 15;
  const int h    = lane >> 4;
  const int n0   = ng * 64;

  const __bf16* arow = abf + (size_t)(mt * 16 + m) * D_DIM;

  Acc8 c[4];
#pragma unroll
  for (int t = 0; t < 4; ++t)
#pragma unroll
    for (int j = 0; j < 8; ++j) c[t].f[j] = 0.0f;

  Frag16 a[2];
  Frag16 bb[2][4];

  auto loadAB = [&](int buf, int k0) {
    a[buf].q[0] = *(const uint4*)(arow + k0 + h * 16);
    a[buf].q[1] = *(const uint4*)(arow + k0 + h * 16 + 8);
#pragma unroll
    for (int t = 0; t < 4; ++t) {
      const __bf16* wrow = wbf + (size_t)(n0 + t * 16 + m) * D_DIM + k0;  // B(k,n)=Wv[n][k]
      bb[buf][t].q[0] = *(const uint4*)(wrow + h * 16);
      bb[buf][t].q[1] = *(const uint4*)(wrow + h * 16 + 8);
    }
  };

  loadAB(0, 0);
  for (int it = 0; it < D_DIM / 32; ++it) {
    const int cur = it & 1;
    if (it + 1 < D_DIM / 32) loadAB(cur ^ 1, (it + 1) * 32);   // prefetch next step
#pragma unroll
    for (int t = 0; t < 4; ++t)
      c[t].v = __builtin_amdgcn_wmma_f32_16x16x32_bf16(false, a[cur].v, false, bb[cur][t].v,
                                                       (short)0, c[t].v, false, false);
  }

#pragma unroll
  for (int t = 0; t < 4; ++t) {
    const int col  = n0 + t * 16 + m;
    const float bi = bv[col];
#pragma unroll
    for (int v = 0; v < 8; ++v) {
      const int row = mt * 16 + v + 8 * h;
      vbf[(size_t)row * D_DIM + col] = (__bf16)tanhf(c[t].f[v] + bi);
    }
  }
}

// ---------------------------------------------------------------------------
// TDM: issue one tensor_load_to_lds for a [rows x 512] bf16 tile (2-D D#,
// groups 2/3 disabled). Wave-level op: EXEC ignored, call from one wave only.
// Descriptor per CDNA5 ISA 8.3/8.4.
// ---------------------------------------------------------------------------
__device__ __forceinline__ void tdm_load_tile(unsigned lds_byte_addr,
                                              const void* gsrc) {
  unsigned long long ga = (unsigned long long)(uintptr_t)gsrc;
  v4u g0;
  g0[0] = 1u;                                        // count=1 (valid), user mode
  g0[1] = lds_byte_addr;                             // lds_addr
  g0[2] = (unsigned)ga;                              // global_addr[31:0]
  g0[3] = ((unsigned)(ga >> 32) & 0x01FFFFFFu)       // global_addr[56:32]
          | (2u << 30);                              // type = 2 ("image")
  v8u g1;
  g1[0] = 0x00010000u;           // workgroup_mask=0, data_size=1 (2B), no opts
  g1[1] = (unsigned)(D_DIM) << 16;                   // tensor_dim0[15:0] @ bits 63:48
  g1[2] = (unsigned)(S_LEN) << 16;                   // dim0 hi=0 | tensor_dim1[15:0] @ 95:80
  g1[3] = (unsigned)(D_DIM) << 16;                   // dim1 hi=0 | tile_dim0 @ 127:112
  g1[4] = (unsigned)KB;                              // tile_dim1=32, tile_dim2=0
  g1[5] = (unsigned)D_DIM;                           // tensor_dim0_stride[31:0]
  g1[6] = 0u;                                        // stride hi | dim1_stride lo
  g1[7] = 0u;
  asm volatile("tensor_load_to_lds %0, %1" :: "s"(g0), "s"(g1) : "memory");
}

__device__ __forceinline__ void tdm_wait() {
#if __has_builtin(__builtin_amdgcn_s_wait_tensorcnt)
  __builtin_amdgcn_s_wait_tensorcnt(0);
#else
  asm volatile("s_wait_tensorcnt 0x0" ::: "memory");
#endif
}

// ---------------------------------------------------------------------------
// Fused flash attention: per batch b and 32 query rows.
// 8 waves = (qg in {0,1} : 16 q rows) x (dq in {0..3} : 128 out cols)
// scores split-K across the 4 dq waves, reduced via LDS.
// Vk tile staged by the Tensor Data Mover; seq_in tile transposed manually
// (overlaps the TDM transfer).
// ---------------------------------------------------------------------------
__global__ __launch_bounds__(256) void attn(const __bf16* __restrict__ vbf,
                                            const __bf16* __restrict__ abf,
                                            float* __restrict__ out) {
  __shared__ __align__(16) __bf16 vk[KB * D_DIM];          // [key][feat]  32KB (TDM dest)
  __shared__ __align__(16) __bf16 sint[D_DIM * KB];        // [feat][key]  32KB (transposed)
  __shared__ __align__(16) float  scoreP[2][4][16][32];    // partial scores 16KB

  const int tid  = threadIdx.x;
  const int lane = tid & 31;
  const int wid  = tid >> 5;
  const int dq   = wid & 3;
  const int qg   = wid >> 2;
  const int b    = blockIdx.y;
  const int q0   = blockIdx.x * QB + qg * 16;
  const int m    = lane & 15;
  const int h    = lane >> 4;
  const size_t boff = (size_t)b * S_LEN * D_DIM;

  // Query fragments for this wave's 128-feature chunk (held for whole kernel)
  Frag16 aq[4];
  {
    const __bf16* qrow = vbf + boff + (size_t)(q0 + m) * D_DIM + dq * 128;
#pragma unroll
    for (int kk = 0; kk < 4; ++kk) {
      aq[kk].q[0] = *(const uint4*)(qrow + kk * 32 + h * 16);
      aq[kk].q[1] = *(const uint4*)(qrow + kk * 32 + h * 16 + 8);
    }
  }

  Acc8 o[8];
#pragma unroll
  for (int t = 0; t < 8; ++t)
#pragma unroll
    for (int j = 0; j < 8; ++j) o[t].f[j] = 0.0f;
  float mrun = -INFINITY, lrun = 0.0f;

  const unsigned vk_lds = (unsigned)(uintptr_t)(&vk[0]);   // LDS byte address

  for (int t0 = 0; t0 < S_LEN; t0 += KB) {
    __syncthreads();                                    // previous-iter LDS reads done

    // ---- TDM: wave 0 DMAs the 32x512 bf16 Vk tile straight into LDS ----
    if (wid == 0) tdm_load_tile(vk_lds, vbf + boff + (size_t)t0 * D_DIM);

    // ---- meanwhile all threads stage seq_in transposed: [feat][key] ----
    {
      const uint4* src2 = (const uint4*)(abf + boff + (size_t)t0 * D_DIM);
#pragma unroll
      for (int i = 0; i < 8; ++i) {
        const int c = i * 256 + tid;
        const int trow = c >> 6;              // key row (512 feats = 64 chunks)
        const int d0c  = (c & 63) * 8;
        U4B u; u.q = src2[c];
#pragma unroll
        for (int j = 0; j < 8; ++j) sint[(d0c + j) * KB + trow] = u.b[j];
      }
    }
    if (wid == 0) tdm_wait();
    __syncthreads();

    // ---- split-K partial scores: this wave covers features [dq*128, dq*128+128) ----
    Acc8 sc[2];
#pragma unroll
    for (int t = 0; t < 2; ++t)
#pragma unroll
      for (int j = 0; j < 8; ++j) sc[t].f[j] = 0.0f;
#pragma unroll
    for (int kk = 0; kk < 4; ++kk) {
#pragma unroll
      for (int t = 0; t < 2; ++t) {
        const __bf16* krow = vk + (size_t)(t * 16 + m) * D_DIM + dq * 128 + kk * 32;
        Frag16 bb;                                       // B(k=feat, n=key)
        bb.q[0] = *(const uint4*)(krow + h * 16);
        bb.q[1] = *(const uint4*)(krow + h * 16 + 8);
        sc[t].v = __builtin_amdgcn_wmma_f32_16x16x32_bf16(false, aq[kk].v, false, bb.v,
                                                          (short)0, sc[t].v, false, false);
      }
    }
#pragma unroll
    for (int t = 0; t < 2; ++t)
#pragma unroll
      for (int v = 0; v < 8; ++v)
        scoreP[qg][dq][v + 8 * h][t * 16 + m] = sc[t].f[v];
    __syncthreads();

    // ---- reduce partials, online softmax (row stats via shfl_xor across halves) ----
    float s[16];
#pragma unroll
    for (int j = 0; j < 16; ++j) s[j] = 0.0f;
#pragma unroll
    for (int w = 0; w < 4; ++w) {
      const float* row = &scoreP[qg][w][m][0];
      float4 r0 = *(const float4*)(row + h * 8);
      float4 r1 = *(const float4*)(row + h * 8 + 4);
      float4 r2 = *(const float4*)(row + 16 + h * 8);
      float4 r3 = *(const float4*)(row + 16 + h * 8 + 4);
      s[0]  += r0.x; s[1]  += r0.y; s[2]  += r0.z; s[3]  += r0.w;
      s[4]  += r1.x; s[5]  += r1.y; s[6]  += r1.z; s[7]  += r1.w;
      s[8]  += r2.x; s[9]  += r2.y; s[10] += r2.z; s[11] += r2.w;
      s[12] += r3.x; s[13] += r3.y; s[14] += r3.z; s[15] += r3.w;
    }
    float lmax = s[0];
#pragma unroll
    for (int j = 1; j < 16; ++j) lmax = fmaxf(lmax, s[j]);
    const float rowmax = fmaxf(lmax, __shfl_xor(lmax, 16, 32));
    const float mnew   = fmaxf(mrun, rowmax);
    const float alpha  = __expf(mrun - mnew);

    Frag16 p;                                            // P in A-fragment order
    float lsum = 0.0f;
#pragma unroll
    for (int j = 0; j < 16; ++j) {
      const float e = __expf(s[j] - mnew);
      p.v[j] = (__bf16)e;
      lsum += e;
    }
    lrun = lrun * alpha + (lsum + __shfl_xor(lsum, 16, 32));
    mrun = mnew;

    float av[8];
#pragma unroll
    for (int v = 0; v < 8; ++v) av[v] = __shfl(alpha, v + 8 * h, 32);
#pragma unroll
    for (int t = 0; t < 8; ++t)
#pragma unroll
      for (int v = 0; v < 8; ++v) o[t].f[v] *= av[v];

    // ---- O += P @ seq_in[keys]   (B from transposed LDS: contiguous per lane) ----
    const int d0 = dq * 128;
#pragma unroll
    for (int t = 0; t < 8; ++t) {
      const __bf16* srow = sint + (size_t)(d0 + t * 16 + m) * KB;
      Frag16 bb;                                         // B(k=key, n=d)
      bb.q[0] = *(const uint4*)(srow + h * 8);
      bb.q[1] = *(const uint4*)(srow + 16 + h * 8);
      o[t].v = __builtin_amdgcn_wmma_f32_16x16x32_bf16(false, p.v, false, bb.v,
                                                       (short)0, o[t].v, false, false);
    }
  }

  // ---- finalize: divide by row sums, store out[S,B,D] ----
  float lv[8];
#pragma unroll
  for (int v = 0; v < 8; ++v) lv[v] = __shfl(lrun, v + 8 * h, 32);
#pragma unroll
  for (int t = 0; t < 8; ++t) {
    const int col = dq * 128 + t * 16 + m;
#pragma unroll
    for (int v = 0; v < 8; ++v) {
      const int row = q0 + v + 8 * h;
      out[((size_t)row * B_SZ + b) * D_DIM + col] = o[t].f[v] / lv[v];
    }
  }
}

// ---------------------------------------------------------------------------
extern "C" void kernel_launch(void* const* d_in, const int* in_sizes, int n_in,
                              void* d_out, int out_size, void* d_ws, size_t ws_size,
                              hipStream_t stream) {
  const float* seq = (const float*)d_in[0];   // [S,B,D]
  const float* Wv  = (const float*)d_in[1];   // [D,D]
  const float* bv  = (const float*)d_in[2];   // [D]
  float* out = (float*)d_out;                 // [S,B,D]

  char* ws = (char*)d_ws;
  __bf16* wbf = (__bf16*)ws;                                    // 512 KB
  __bf16* abf = (__bf16*)(ws + (size_t)(1 << 19));              // 32 MB  seq_in bf16
  __bf16* vbf = (__bf16*)(ws + (size_t)(1 << 19) + ((size_t)32 << 20)); // 32 MB value bf16

  const size_t total = (size_t)S_LEN * B_SZ * D_DIM;
  cvt_seq<<<(unsigned)((total + 255) / 256), 256, 0, stream>>>(seq, abf);
  cvt_w<<<(D_DIM * D_DIM + 255) / 256, 256, 0, stream>>>(Wv, wbf, D_DIM * D_DIM);

  // 2048 m-tiles * 8 n-groups = 16384 waves / 4 waves-per-block
  value_gemm<<<4096, 128, 0, stream>>>(abf, wbf, bv, vbf);

  attn<<<dim3(S_LEN / QB, B_SZ), 256, 0, stream>>>(vbf, abf, out);
}
